// SSMLayer_79156247266069
// MI455X (gfx1250) — matrix operand order repaired and checked by reading, not compiled, not run on hardware
//
#include <hip/hip_runtime.h>
#include <math.h>

// Problem dimensions (fixed by the reference)
#define BATCH   8
#define SEQ     4096
#define DMODEL  1024
#define DINNER  2048
#define NST     16
#define MROWS   (BATCH*SEQ)       // 32768 rows
#define KIN     DMODEL            // 1024
#define CHUNK   512
#define NCHUNK  (SEQ/CHUNK)       // 8

typedef float  v2f  __attribute__((ext_vector_type(2)));
typedef float  v8f  __attribute__((ext_vector_type(8)));
typedef __bf16 bf16;
typedef __bf16 v8bf  __attribute__((ext_vector_type(8)));
typedef __bf16 v16bf __attribute__((ext_vector_type(16)));

// GCC-style vector to match the async-to-LDS builtin's parameter type exactly
typedef int v4i_vs __attribute__((vector_size(16)));
typedef __attribute__((address_space(1))) v4i_vs* as1_v4i_p;
typedef __attribute__((address_space(3))) v4i_vs* as3_v4i_p;

#if defined(__has_builtin)
#  if __has_builtin(__builtin_amdgcn_global_load_async_to_lds_b128)
#    define HAVE_ASYNC 1
#  else
#    define HAVE_ASYNC 0
#  endif
#else
#  define HAVE_ASYNC 0
#endif

// D = A(16x4)*B(4x16)+C, f32
__device__ __forceinline__ v8f wmma4(v2f a, v2f b, v8f c) {
  return __builtin_amdgcn_wmma_f32_16x16x4_f32(false, a, false, b, (short)0, c,
                                               false, false);
}
// D = A(16x32)*B(32x16)+C, bf16 in / f32 acc
__device__ __forceinline__ v8f wmma_bf(v16bf a, v16bf b, v8f c) {
  return __builtin_amdgcn_wmma_f32_16x16x32_bf16(false, a, false, b, (short)0, c,
                                                 false, false);
}

__device__ __forceinline__ float silu_f(float x) {
  return x / (1.0f + __expf(-x));
}

// split 8 consecutive f32 into bf16 hi + bf16 lo correction
__device__ __forceinline__ void cvt_split8(const float4 a, const float4 b,
                                           v8bf& hi, v8bf& lo) {
  float v[8] = {a.x, a.y, a.z, a.w, b.x, b.y, b.z, b.w};
#pragma unroll
  for (int i = 0; i < 8; i++) {
    bf16 h = (bf16)v[i];
    hi[i] = h;
    lo[i] = (bf16)(v[i] - (float)h);
  }
}

// ---------------- prep kernels ----------------

// xpwT[c][i] = xp_w[i][c]  (DINNER x 32)
__global__ void k_transpose_xpw(const float* __restrict__ w, float* __restrict__ wT) {
  int idx = blockIdx.x * 256 + threadIdx.x;       // DINNER*32 threads
  int c = idx / 32, i = idx % 32;
  wT[idx] = w[(size_t)i * DINNER + c];
}

// owT[n][d] = out_w[d][n]  (NST x DMODEL)
__global__ void k_transpose_outw(const float* __restrict__ w, float* __restrict__ wT) {
  int idx = blockIdx.x * 256 + threadIdx.x;       // NST*DMODEL threads
  int n = idx / DMODEL, d = idx % DMODEL;
  wT[idx] = w[(size_t)d * NST + n];
}

__global__ void k_zero(float* __restrict__ p, size_t n) {
  size_t i = (size_t)blockIdx.x * blockDim.x + threadIdx.x;
  size_t stride = (size_t)gridDim.x * blockDim.x;
  for (; i < n; i += stride) p[i] = 0.0f;
}

// ---------------- kernel 1: in_proj GEMM  h0 = x @ in_w.T + in_b ----------------
// Split-bf16: x ~= x_hi + x_lo, w ~= w_hi + w_lo (bf16); acc f32 of
// hi*hi + hi*lo + lo*hi via V_WMMA_F32_16X16X32_BF16 (8x K per instr vs f32 WMMA).
// Block: 256 threads = 8 waves (2x4). Block tile 64x128, wave tile 32x32.
#define BM  64
#define BN  128
#define KC2 32

__global__ __launch_bounds__(256) void k_inproj(const float* __restrict__ x,
                                                const float* __restrict__ in_w,
                                                const float* __restrict__ in_b,
                                                float* __restrict__ h0) {
  // A tiles [m][k], B tiles [n][k] (== native in_w layout, no transpose needed)
  __shared__ __attribute__((aligned(16))) bf16 Ahi[BM * KC2];
  __shared__ __attribute__((aligned(16))) bf16 Alo[BM * KC2];
  __shared__ __attribute__((aligned(16))) bf16 Bhi[BN * KC2];
  __shared__ __attribute__((aligned(16))) bf16 Blo[BN * KC2];

  const int tid   = threadIdx.x;
  const int lane  = tid & 31;
  const int wave  = tid >> 5;
  const int waveM = wave >> 2;   // 0..1
  const int waveN = wave & 3;    // 0..3
  const int lm    = lane & 15;
  const int lh    = lane >> 4;   // 0/1
  const int m0    = blockIdx.x * BM;
  const int n0    = blockIdx.y * BN;

  const v8f vzero = {0.f, 0.f, 0.f, 0.f, 0.f, 0.f, 0.f, 0.f};
  v8f acc[2][2] = {{vzero, vzero}, {vzero, vzero}};

  // staging assignments
  const int ar = tid >> 2;             // A row 0..63
  const int ac = (tid & 3) * 8;        // A col {0,8,16,24}
  const int brow = tid >> 1;           // B row (n) 0..127
  const int bcol = (tid & 1) * 16;     // B col {0,16}

  const float* xA = x + (size_t)(m0 + ar) * KIN + ac;
  const float* wB = in_w + (size_t)(n0 + brow) * KIN + bcol;

  for (int kc0 = 0; kc0 < KIN; kc0 += KC2) {
    {  // stage A (64x32 f32 -> bf16 hi/lo)
      const float4 a0 = *(const float4*)(xA + kc0);
      const float4 a1 = *(const float4*)(xA + kc0 + 4);
      v8bf hi, lo;
      cvt_split8(a0, a1, hi, lo);
      *(v8bf*)(Ahi + ar * KC2 + ac) = hi;
      *(v8bf*)(Alo + ar * KC2 + ac) = lo;
    }
    {  // stage B (128x32 f32 -> bf16 hi/lo), coalesced along K
      const float4 b0 = *(const float4*)(wB + kc0);
      const float4 b1 = *(const float4*)(wB + kc0 + 4);
      const float4 b2 = *(const float4*)(wB + kc0 + 8);
      const float4 b3 = *(const float4*)(wB + kc0 + 12);
      v8bf hi0, lo0, hi1, lo1;
      cvt_split8(b0, b1, hi0, lo0);
      cvt_split8(b2, b3, hi1, lo1);
      *(v8bf*)(Bhi + brow * KC2 + bcol)     = hi0;
      *(v8bf*)(Blo + brow * KC2 + bcol)     = lo0;
      *(v8bf*)(Bhi + brow * KC2 + bcol + 8) = hi1;
      *(v8bf*)(Blo + brow * KC2 + bcol + 8) = lo1;
    }
    if (kc0 + KC2 < KIN) {  // hint next chunk into cache
      __builtin_prefetch(xA + kc0 + KC2, 0, 1);
      __builtin_prefetch(wB + kc0 + KC2, 0, 1);
    }
    __syncthreads();

    // fragments: A lane holds m=lm, K = [kb..kb+8) U [kb+16..kb+24), kb = lh*8
    //            B lane holds n=lm, K = [kb..kb+8) U [kb+16..kb+24)
    const int kb   = lh * 8;
    union ubf { v16bf v; v8bf h[2]; };

    ubf ah[2], al[2], bh[2], bl[2];
#pragma unroll
    for (int tm = 0; tm < 2; tm++) {
      const int m = (waveM * 32 + tm * 16 + lm) * KC2;
      ah[tm].h[0] = *(const v8bf*)(Ahi + m + kb);
      ah[tm].h[1] = *(const v8bf*)(Ahi + m + kb + 16);
      al[tm].h[0] = *(const v8bf*)(Alo + m + kb);
      al[tm].h[1] = *(const v8bf*)(Alo + m + kb + 16);
    }
#pragma unroll
    for (int tn = 0; tn < 2; tn++) {
      const int n = (waveN * 32 + tn * 16 + lm) * KC2;
      bh[tn].h[0] = *(const v8bf*)(Bhi + n + kb);
      bh[tn].h[1] = *(const v8bf*)(Bhi + n + kb + 16);
      bl[tn].h[0] = *(const v8bf*)(Blo + n + kb);
      bl[tn].h[1] = *(const v8bf*)(Blo + n + kb + 16);
    }
#pragma unroll
    for (int tm = 0; tm < 2; tm++)
#pragma unroll
      for (int tn = 0; tn < 2; tn++) {
        acc[tm][tn] = wmma_bf(ah[tm].v, bh[tn].v, acc[tm][tn]);
        acc[tm][tn] = wmma_bf(ah[tm].v, bl[tn].v, acc[tm][tn]);
        acc[tm][tn] = wmma_bf(al[tm].v, bh[tn].v, acc[tm][tn]);
      }
    __syncthreads();
  }

#pragma unroll
  for (int tm = 0; tm < 2; tm++)
#pragma unroll
    for (int tn = 0; tn < 2; tn++) {
      const int gm = m0 + waveM * 32 + tm * 16 + lh * 8;
      const int gn = n0 + waveN * 32 + tn * 16 + lm;
      const float bias = in_b[gn];
#pragma unroll
      for (int r = 0; r < 8; r++)
        h0[(size_t)(gm + r) * DINNER + gn] = acc[tm][tn][r] + bias;
    }
}

// ---------------- kernel 2: conv + SiLU + rowsum + x_proj partial ----------------
// Block handles TS=64 seq positions x TC=64 channels of one batch; partial xp /
// rowsum combined across channel chunks with global f32 atomics. h is never
// written to HBM. Halo tile staged with GLOBAL_LOAD_ASYNC_TO_LDS when available.
#define TS 64
#define TC 64

__global__ __launch_bounds__(256) void k_conv_xp(const float* __restrict__ h0,
                                                 const float* __restrict__ conv_w,
                                                 const float* __restrict__ conv_b,
                                                 const float* __restrict__ xpwT,
                                                 float* __restrict__ xp,
                                                 float* __restrict__ rsum) {
  __shared__ __attribute__((aligned(16))) float hin[(TS + 3) * TC];
  __shared__ __attribute__((aligned(16))) float hconv[TS * TC];
  __shared__ float rs[TS];
  const int tid = threadIdx.x;
  const int b   = blockIdx.z;
  const int s0  = blockIdx.x * TS;
  const int c0  = blockIdx.y * TC;

#if HAVE_ASYNC
  for (int i = tid; i < (TS + 3) * (TC / 4); i += 256) {
    const int r  = i / (TC / 4);
    const int c4 = (i % (TC / 4)) * 4;
    const int gs = s0 - 3 + r;  // halo of 3 on the left (causal pad)
    if (gs >= 0) {
      __builtin_amdgcn_global_load_async_to_lds_b128(
          (as1_v4i_p)(h0 + (size_t)(b * SEQ + gs) * DINNER + c0 + c4),
          (as3_v4i_p)(hin + r * TC + c4), 0, 0);
    } else {
      *(float4*)(hin + r * TC + c4) = make_float4(0.f, 0.f, 0.f, 0.f);
    }
  }
#  if __has_builtin(__builtin_amdgcn_s_wait_asynccnt)
  __builtin_amdgcn_s_wait_asynccnt(0);
#  else
  asm volatile("s_wait_asynccnt 0x0" ::: "memory");
#  endif
#else
  for (int i = tid; i < (TS + 3) * TC; i += 256) {
    const int r = i / TC, c = i % TC;
    const int gs = s0 - 3 + r;
    hin[i] = (gs >= 0) ? h0[(size_t)(b * SEQ + gs) * DINNER + c0 + c] : 0.0f;
  }
#endif
  if (tid < TS) rs[tid] = 0.0f;
  __syncthreads();

  {
    const int c  = tid & 63;
    const int sb = (tid >> 6) * 16;
    const int gc = c0 + c;
    const float w0 = conv_w[gc * 4 + 0], w1 = conv_w[gc * 4 + 1];
    const float w2 = conv_w[gc * 4 + 2], w3 = conv_w[gc * 4 + 3];
    const float cb = conv_b[gc];
#pragma unroll
    for (int j = 0; j < 16; j++) {
      const int s = sb + j;
      float v = cb + w0 * hin[s * TC + c] + w1 * hin[(s + 1) * TC + c] +
                w2 * hin[(s + 2) * TC + c] + w3 * hin[(s + 3) * TC + c];
      v = silu_f(v);
      hconv[s * TC + c] = v;
      atomicAdd(&rs[s], v);   // ds_add_f32
    }
  }
  __syncthreads();

  if (tid < TS) atomicAdd(&rsum[b * SEQ + s0 + tid], rs[tid]);

  // xp partial: hconv[64x64] @ xpwT_chunk[64x32] via f32 WMMA; 4 Mtiles x 2 Ntiles
  const int lane = tid & 31, wave = tid >> 5;
  const int mt = wave >> 1, nt = wave & 1;
  const int lm = lane & 15, lh = lane >> 4;
  v8f acc = {0.f, 0.f, 0.f, 0.f, 0.f, 0.f, 0.f, 0.f};
#pragma unroll
  for (int kk = 0; kk < TC; kk += 4) {
    const int k0 = kk + 2 * lh;
    const v2f a = *(const v2f*)(hconv + (mt * 16 + lm) * TC + k0);
    const v2f bf = {xpwT[(size_t)(c0 + k0) * 32 + nt * 16 + lm],
                    xpwT[(size_t)(c0 + k0 + 1) * 32 + nt * 16 + lm]};
    acc = wmma4(a, bf, acc);
  }
  const int gm = b * SEQ + s0 + mt * 16 + lh * 8;
  const int gn = nt * 16 + lm;
#pragma unroll
  for (int r = 0; r < 8; r++)
    atomicAdd(&xp[(size_t)(gm + r) * 32 + gn], acc[r]);
}

// ---------------- kernel 3: gates ----------------
__global__ void k_gates(const float* __restrict__ xp, const float* __restrict__ xp_b,
                        const float* __restrict__ rsum, float* __restrict__ Aarr,
                        float* __restrict__ Barr) {
  const int idx = blockIdx.x * 256 + threadIdx.x;  // MROWS*16 threads
  const int m = idx >> 4, i = idx & 15;
  const float av = xp[(size_t)m * 32 + i] + xp_b[i];
  const float bv = xp[(size_t)m * 32 + 16 + i] + xp_b[16 + i];
  Aarr[idx] = __expf(-__expf(av));
  Barr[idx] = silu_f(bv) * rsum[m];
}

// ---------------- kernel 4: chunked scan (3 phases) ----------------
__global__ void k_scan_chunks(const float* __restrict__ Aarr,
                              const float* __restrict__ Barr,
                              float* __restrict__ P, float* __restrict__ Q) {
  const int t = blockIdx.x * 256 + threadIdx.x;    // BATCH*NCHUNK*16 = 1024
  const int b = t / (NCHUNK * 16);
  const int c = (t >> 4) % NCHUNK;
  const int n = t & 15;
  float s = 0.f, p = 1.f;
  const size_t base = (size_t)(b * SEQ + c * CHUNK) * 16 + n;
  for (int j = 0; j < CHUNK; j++) {
    const float a = Aarr[base + (size_t)j * 16];
    s = s * a + Barr[base + (size_t)j * 16];
    p *= a;
  }
  P[(b * NCHUNK + c) * 16 + n] = p;
  Q[(b * NCHUNK + c) * 16 + n] = s;
}

__global__ void k_scan_carry(const float* __restrict__ P, const float* __restrict__ Q,
                             float* __restrict__ initS, float* __restrict__ finalState) {
  const int t = threadIdx.x;  // 128 threads
  const int b = t >> 4, n = t & 15;
  float carry = 0.f;
  for (int c = 0; c < NCHUNK; c++) {
    const int i = (b * NCHUNK + c) * 16 + n;
    initS[i] = carry;
    carry = carry * P[i] + Q[i];
  }
  finalState[b * 16 + n] = carry;
}

__global__ void k_scan_apply(const float* __restrict__ Aarr,
                             const float* __restrict__ Barr,
                             const float* __restrict__ initS, float* __restrict__ ys) {
  const int t = blockIdx.x * 256 + threadIdx.x;    // BATCH*NCHUNK*16
  const int b = t / (NCHUNK * 16);
  const int c = (t >> 4) % NCHUNK;
  const int n = t & 15;
  float s = initS[(b * NCHUNK + c) * 16 + n];
  const size_t base = (size_t)(b * SEQ + c * CHUNK) * 16 + n;
  for (int j = 0; j < CHUNK; j++) {
    const size_t idx = base + (size_t)j * 16;
    s = s * Aarr[idx] + Barr[idx];
    ys[idx] = s;
  }
}

// ---------------- kernel 5: out_proj  out = ys @ out_w.T + out_b ----------------
// K=16: each wave keeps its 4 A fragments in registers and sweeps all 64 N tiles.
__global__ __launch_bounds__(256) void k_outproj(const float* __restrict__ ys,
                                                 const float* __restrict__ owT,
                                                 const float* __restrict__ out_b,
                                                 float* __restrict__ out) {
  const int lane = threadIdx.x & 31, wave = threadIdx.x >> 5;
  const int lm = lane & 15, lh = lane >> 4;
  const int m0 = (blockIdx.x * 8 + wave) * 16;
  v2f a[4];
#pragma unroll
  for (int kk = 0; kk < 4; kk++) {
    const int k0 = kk * 4 + 2 * lh;
    a[kk] = *(const v2f*)(ys + (size_t)(m0 + lm) * 16 + k0);
  }
  for (int nt = 0; nt < DMODEL / 16; nt++) {
    const int n = nt * 16 + lm;
    v8f acc = {0.f, 0.f, 0.f, 0.f, 0.f, 0.f, 0.f, 0.f};
#pragma unroll
    for (int kk = 0; kk < 4; kk++) {
      const int k0 = kk * 4 + 2 * lh;
      const v2f bf = {owT[(size_t)k0 * DMODEL + n],
                      owT[(size_t)(k0 + 1) * DMODEL + n]};
      acc = wmma4(a[kk], bf, acc);
    }
    const float bias = out_b[n];
#pragma unroll
    for (int r = 0; r < 8; r++)
      out[(size_t)(m0 + lh * 8 + r) * DMODEL + n] = acc[r] + bias;
  }
}

// ---------------- launch ----------------
extern "C" void kernel_launch(void* const* d_in, const int* in_sizes, int n_in,
                              void* d_out, int out_size, void* d_ws, size_t ws_size,
                              hipStream_t stream) {
  const float* x      = (const float*)d_in[0];
  const float* in_w   = (const float*)d_in[1];
  const float* in_b   = (const float*)d_in[2];
  const float* conv_w = (const float*)d_in[3];
  const float* conv_b = (const float*)d_in[4];
  const float* xp_w   = (const float*)d_in[5];
  const float* xp_b   = (const float*)d_in[6];
  const float* out_w  = (const float*)d_in[7];
  const float* out_b  = (const float*)d_in[8];
  float* out = (float*)d_out;

  float* ws = (float*)d_ws;
  size_t off = 0;
  float* xpwT  = ws + off; off += (size_t)DINNER * 32;      // 64K
  float* owT   = ws + off; off += (size_t)NST * DMODEL;     // 16K
  float* h0    = ws + off; off += (size_t)MROWS * DINNER;   // 64M (256 MiB)
  float* xp    = ws + off; off += (size_t)MROWS * 32;       // atomic acc
  float* rsum  = ws + off; off += (size_t)MROWS;            // atomic acc (contiguous w/ xp)
  float* Aarr  = ws + off; off += (size_t)MROWS * 16;
  float* Barr  = ws + off; off += (size_t)MROWS * 16;
  float* ysb   = ws + off; off += (size_t)MROWS * 16;
  float* P     = ws + off; off += (size_t)BATCH * NCHUNK * 16;
  float* Q     = ws + off; off += (size_t)BATCH * NCHUNK * 16;
  float* initS = ws + off; off += (size_t)BATCH * NCHUNK * 16;

  // prep: transposed small weights + zero the atomic accumulators (every launch)
  k_transpose_xpw <<<(DINNER * 32) / 256,  256, 0, stream>>>(xp_w, xpwT);
  k_transpose_outw<<<(NST * DMODEL) / 256, 256, 0, stream>>>(out_w, owT);
  k_zero<<<256, 256, 0, stream>>>(xp, (size_t)MROWS * 32 + MROWS);  // xp + rsum

  dim3 g1(MROWS / BM, DINNER / BN);
  k_inproj<<<g1, 256, 0, stream>>>(x, in_w, in_b, h0);

  dim3 g2(SEQ / TS, DINNER / TC, BATCH);
  k_conv_xp<<<g2, 256, 0, stream>>>(h0, conv_w, conv_b, xpwT, xp, rsum);

  k_gates<<<(MROWS * 16) / 256, 256, 0, stream>>>(xp, xp_b, rsum, Aarr, Barr);

  k_scan_chunks<<<(BATCH * NCHUNK * 16) / 256, 256, 0, stream>>>(Aarr, Barr, P, Q);
  k_scan_carry<<<1, 128, 0, stream>>>(P, Q, initS, out + (size_t)MROWS * DMODEL);
  k_scan_apply<<<(BATCH * NCHUNK * 16) / 256, 256, 0, stream>>>(Aarr, Barr, initS, ysb);

  k_outproj<<<MROWS / (16 * 8), 256, 0, stream>>>(ysb, owT, out_b, out);
}
